// MultiHeadSelfAttention_31318901522838
// MI455X (gfx1250) — matrix operand order repaired and checked by reading, not compiled, not run on hardware
//
#include <hip/hip_runtime.h>
#include <hip/hip_bf16.h>
#include <math.h>

// ---------------------------------------------------------------------------
// MHSA  B=8, N=1024, C=768, H=12, D=64  on gfx1250 (wave32, WMMA, TDM).
// Pipeline:
//   1) cast x -> f16; cast+transpose Wqkv, Wproj -> f16 [n][k]
//   2) qkv GEMM (WMMA f16->f32): q,k stored [row][1536]; v stored transposed
//      per head as vT[b,h][d][n]  -> attention V staging needs no transpose
//   3) flash attention (WMMA, online softmax), K/V tiles staged by TDM
//   4) proj GEMM (WMMA f16->f32) -> d_out
// All LDS tile staging is a straight 2D strided copy -> tensor_load_to_lds
// (double buffered, TENSORcnt) when the builtin exists; manual copy fallback.
// ---------------------------------------------------------------------------

typedef _Float16 v16h __attribute__((ext_vector_type(16)));
typedef _Float16 v8h  __attribute__((ext_vector_type(8)));
typedef float    v8f  __attribute__((ext_vector_type(8)));
typedef unsigned int u32x4 __attribute__((ext_vector_type(4)));
typedef int          i32x4 __attribute__((ext_vector_type(4)));
typedef int          i32x8 __attribute__((ext_vector_type(8)));

#if __has_builtin(__builtin_amdgcn_tensor_load_to_lds)
#define HAS_TDM 1
#endif

#if __has_builtin(__builtin_amdgcn_s_wait_tensorcnt)
#define WAIT_TENSORCNT(n) __builtin_amdgcn_s_wait_tensorcnt(n)
#else
#define WAIT_TENSORCNT(n) asm volatile("s_wait_tensorcnt " #n)
#endif

static __device__ __forceinline__ v16h frag16(const _Float16* p0, const _Float16* p1) {
    v8h a = *(const v8h*)p0;
    v8h b = *(const v8h*)p1;
    return __builtin_shufflevector(a, b, 0,1,2,3,4,5,6,7,8,9,10,11,12,13,14,15);
}

static __device__ __forceinline__ v8f wmma_f16(v16h a, v16h b, v8f c) {
    return __builtin_amdgcn_wmma_f32_16x16x32_f16(false, a, false, b, (short)0, c, false, false);
}

static __device__ __forceinline__ float redmax16(float v) {
    #pragma unroll
    for (int m = 1; m <= 8; m <<= 1) v = fmaxf(v, __shfl_xor(v, m, 32));
    return v;
}
static __device__ __forceinline__ float redsum16(float v) {
    #pragma unroll
    for (int m = 1; m <= 8; m <<= 1) v += __shfl_xor(v, m, 32);
    return v;
}

#ifdef HAS_TDM
// 2D tile load: `height` rows of `width` f16 elements, global row stride
// `stride_elems`, into contiguous LDS at byte offset lds_off. D# per ISA
// CDNA5 ISA (group0: count/type/addrs, group1: dims).
// 6-arg builtin form: (u32x4 g0, i32x8 g1, i32x4 g2, i32x4 g3, i32x8, i32 cpol)
static __device__ __forceinline__ void tdm_load_2d(unsigned lds_off, const void* gaddr,
                                                   int width, int height, long stride_elems) {
    unsigned long long ga = (unsigned long long)gaddr;
    u32x4 g0;
    g0[0] = 1u;                                   // count=1, user descriptor
    g0[1] = lds_off;                              // lds_addr (bytes)
    g0[2] = (unsigned)(ga & 0xffffffffu);         // global_addr[31:0]
    g0[3] = (unsigned)((ga >> 32) & 0x1ffffffu)   // global_addr[56:32]
          | 0x80000000u;                          // type=2 ("image")
    unsigned w = (unsigned)width, hgt = (unsigned)height;
    unsigned long long st = (unsigned long long)stride_elems;
    i32x8 g1;
    g1[0] = (int)(1u << 16);                      // data_size=1 (2 bytes)
    g1[1] = (int)((w & 0xffffu) << 16);           // tensor_dim0[15:0]
    g1[2] = (int)((w >> 16) | ((hgt & 0xffffu) << 16));   // td0 hi | td1 lo
    g1[3] = (int)((hgt >> 16) | ((w & 0xffffu) << 16));   // td1 hi | tile_dim0
    g1[4] = (int)(hgt & 0xffffu);                 // tile_dim1 | tile_dim2=0
    g1[5] = (int)(st & 0xffffffffu);              // tensor_dim0_stride lo
    g1[6] = (int)((st >> 32) & 0xffffu);          // stride hi | dim1_stride lo
    g1[7] = 0;
    i32x4 z4 = {0, 0, 0, 0};
    i32x8 z8 = {0, 0, 0, 0, 0, 0, 0, 0};
    __builtin_amdgcn_tensor_load_to_lds(g0, g1, z4, z4, z8, 0);
}
#endif

// ---------------------------------------------------------------------------
// elementwise fp32 -> f16
// ---------------------------------------------------------------------------
__global__ __launch_bounds__(256) void cvt_f32_f16(const float* __restrict__ in,
                                                   _Float16* __restrict__ out, int n) {
    int i = blockIdx.x * 256 + threadIdx.x;
    if (i < n) out[i] = (_Float16)in[i];
}

// in[R][N] fp32 -> out[N][R] f16  (one-time weight transpose)
__global__ __launch_bounds__(256) void cvt_transpose_f16(const float* __restrict__ in,
                                                         _Float16* __restrict__ out,
                                                         int R, int N) {
    int i = blockIdx.x * 256 + threadIdx.x;
    if (i < R * N) {
        int r = i / N, n = i - r * N;
        out[(size_t)n * R + r] = (_Float16)in[i];
    }
}

// ---------------------------------------------------------------------------
// WMMA GEMM:  C[M,N] = A[M,K] @ BT[N,K]^T + bias[N]
// 256 threads / 8 waves; block tile 128x128; K staged 64/iter, double buffered.
// QKV mode: f16 out, cols [0,1536) -> qk[row][1536], cols [1536,2304) ->
//           vT[(b*12+h)*64+d][1024]  (n = row%1024).
// plain mode: f32 out to Cout with stride N.
// ---------------------------------------------------------------------------
template <bool QKV>
__global__ __launch_bounds__(256) void gemm_wmma(const _Float16* __restrict__ A,
                                                 const _Float16* __restrict__ BT,
                                                 const float* __restrict__ bias,
                                                 void* __restrict__ Cout,
                                                 _Float16* __restrict__ vT,
                                                 int M, int N, int K) {
    __shared__ _Float16 lA[2][128 * 64];   // [m][k]
    __shared__ _Float16 lB[2][128 * 64];   // [n][k]

    const int t    = threadIdx.x;
    const int lane = t & 31;
    const int wave = t >> 5;
    const int wy   = wave >> 1;
    const int wx   = wave & 1;
    const int hi   = lane >> 4;
    const int ln   = lane & 15;

    const int Mbase = blockIdx.y * 128;
    const int Nbase = blockIdx.x * 128;
    const int nk    = K >> 6;

    v8f acc[2][4];
    #pragma unroll
    for (int mt = 0; mt < 2; ++mt)
        #pragma unroll
        for (int nt = 0; nt < 4; ++nt) acc[mt][nt] = (v8f)0.0f;

#ifdef HAS_TDM
    if (wave == 0) {
        tdm_load_2d((unsigned)(size_t)&lA[0][0], &A[(size_t)Mbase * K],       64, 128, K);
        tdm_load_2d((unsigned)(size_t)&lB[0][0], &BT[(size_t)Nbase * K],      64, 128, K);
        tdm_load_2d((unsigned)(size_t)&lA[1][0], &A[(size_t)Mbase * K + 64],  64, 128, K);
        tdm_load_2d((unsigned)(size_t)&lB[1][0], &BT[(size_t)Nbase * K + 64], 64, 128, K);
    }
#endif

    for (int kt = 0; kt < nk; ++kt) {
        const int buf = kt & 1;
#ifdef HAS_TDM
        if (wave == 0) {
            if (kt < nk - 1) { WAIT_TENSORCNT(2); } else { WAIT_TENSORCNT(0); }
        }
        __syncthreads();
#else
        __syncthreads();
        #pragma unroll
        for (int i = 0; i < 4; ++i) {           // 1024 8-half chunks per tile
            int c = t + i * 256;
            int row = c >> 3, ko = (c & 7) * 8;
            *(v8h*)&lA[buf][row * 64 + ko] =
                *(const v8h*)&A[(size_t)(Mbase + row) * K + kt * 64 + ko];
            *(v8h*)&lB[buf][row * 64 + ko] =
                *(const v8h*)&BT[(size_t)(Nbase + row) * K + kt * 64 + ko];
        }
        __syncthreads();
#endif
        // ---- 16 WMMAs on the 64-deep K tile ----
        #pragma unroll
        for (int s = 0; s < 2; ++s) {
            v16h fa[2], fb[4];
            #pragma unroll
            for (int mt = 0; mt < 2; ++mt) {
                const _Float16* p = &lA[buf][(wy * 32 + mt * 16 + ln) * 64 + s * 32 + hi * 8];
                fa[mt] = frag16(p, p + 16);
            }
            #pragma unroll
            for (int nt = 0; nt < 4; ++nt) {
                const _Float16* p = &lB[buf][(wx * 64 + nt * 16 + ln) * 64 + s * 32 + hi * 8];
                fb[nt] = frag16(p, p + 16);
            }
            #pragma unroll
            for (int mt = 0; mt < 2; ++mt)
                #pragma unroll
                for (int nt = 0; nt < 4; ++nt)
                    acc[mt][nt] = wmma_f16(fa[mt], fb[nt], acc[mt][nt]);
        }
#ifdef HAS_TDM
        __syncthreads();
        if (wave == 0 && kt + 2 < nk) {
            tdm_load_2d((unsigned)(size_t)&lA[buf][0],
                        &A[(size_t)Mbase * K + (kt + 2) * 64], 64, 128, K);
            tdm_load_2d((unsigned)(size_t)&lB[buf][0],
                        &BT[(size_t)Nbase * K + (kt + 2) * 64], 64, 128, K);
        }
#endif
    }

    // ---- epilogue ----
    #pragma unroll
    for (int mt = 0; mt < 2; ++mt) {
        #pragma unroll
        for (int nt = 0; nt < 4; ++nt) {
            int col = Nbase + wx * 64 + nt * 16 + ln;
            float bv = bias[col];
            #pragma unroll
            for (int j = 0; j < 8; ++j) {
                int row = Mbase + wy * 32 + mt * 16 + hi * 8 + j;
                float v = acc[mt][nt][j] + bv;
                if (QKV) {
                    if (Nbase < 1536) {          // q,k slab, uniform per block
                        ((_Float16*)Cout)[(size_t)row * 1536 + col] = (_Float16)v;
                    } else {                     // v slab -> transposed per head
                        int c2 = col - 1536, h = c2 >> 6, d = c2 & 63;
                        int b = row >> 10, n = row & 1023;
                        vT[((size_t)(b * 12 + h) * 64 + d) * 1024 + n] = (_Float16)v;
                    }
                } else {
                    ((float*)Cout)[(size_t)row * N + col] = v;
                }
            }
        }
    }
}

// ---------------------------------------------------------------------------
// Flash attention: one workgroup per (64 Q rows, b, h); 4 waves x 16 rows.
// qk[row][1536]: q at +h*64, k at +768+h*64.  vT[(b*12+h)*64+d][1024].
// K and V tiles staged by TDM (double buffered).
// ---------------------------------------------------------------------------
__global__ __launch_bounds__(128) void attn_flash(const _Float16* __restrict__ qk,
                                                  const _Float16* __restrict__ vT,
                                                  _Float16* __restrict__ attnout) {
    __shared__ _Float16 lK[2][64 * 64];    // [key][d]
    __shared__ _Float16 lV[2][64 * 64];    // [d][key]
    __shared__ _Float16 lP[4][16 * 64];    // per-wave P scratch [row][key]

    const int t    = threadIdx.x;
    const int lane = t & 31;
    const int w    = t >> 5;
    const int hi   = lane >> 4;
    const int ln   = lane & 15;

    const int r0 = blockIdx.x * 64;
    const int b  = blockIdx.y / 12;
    const int h  = blockIdx.y % 12;
    const float scale = 0.125f;            // 1/sqrt(64)

    const _Float16* kbase = qk + 768 + h * 64;                      // + row*1536
    const _Float16* vbase = vT + (size_t)(b * 12 + h) * 64 * 1024;  // [d][1024]

    // Q fragments (A layout), straight from global
    v16h qa[2];
    {
        int row = r0 + w * 16 + ln;
        const _Float16* qp = qk + (size_t)(b * 1024 + row) * 1536 + h * 64;
        #pragma unroll
        for (int s = 0; s < 2; ++s) {
            const _Float16* p = qp + s * 32 + hi * 8;
            qa[s] = frag16(p, p + 16);
        }
    }

    v8f oacc[4];
    #pragma unroll
    for (int nt = 0; nt < 4; ++nt) oacc[nt] = (v8f)0.0f;
    float rowmax[8], rowsum[8];
    #pragma unroll
    for (int j = 0; j < 8; ++j) { rowmax[j] = -1e30f; rowsum[j] = 0.0f; }

    const int nk = 16;
#ifdef HAS_TDM
    if (w == 0) {
        tdm_load_2d((unsigned)(size_t)&lK[0][0], kbase + (size_t)(b * 1024) * 1536,        64, 64, 1536);
        tdm_load_2d((unsigned)(size_t)&lV[0][0], vbase,                                    64, 64, 1024);
        tdm_load_2d((unsigned)(size_t)&lK[1][0], kbase + (size_t)(b * 1024 + 64) * 1536,   64, 64, 1536);
        tdm_load_2d((unsigned)(size_t)&lV[1][0], vbase + 64,                               64, 64, 1024);
    }
#endif

    for (int kt = 0; kt < nk; ++kt) {
        const int buf = kt & 1;
#ifdef HAS_TDM
        if (w == 0) {
            if (kt < nk - 1) { WAIT_TENSORCNT(2); } else { WAIT_TENSORCNT(0); }
        }
        __syncthreads();
#else
        __syncthreads();
        #pragma unroll
        for (int i = 0; i < 4; ++i) {           // 512 8-half chunks per tile
            int c = t + i * 128;
            int row = c >> 3, ko = (c & 7) * 8;
            *(v8h*)&lK[buf][row * 64 + ko] =
                *(const v8h*)&kbase[(size_t)(b * 1024 + kt * 64 + row) * 1536 + ko];
            *(v8h*)&lV[buf][row * 64 + ko] =
                *(const v8h*)&vbase[(size_t)row * 1024 + kt * 64 + ko];
        }
        __syncthreads();
#endif
        // ---- S = scale * Q @ K^T ----
        v8f sacc[4];
        #pragma unroll
        for (int nt = 0; nt < 4; ++nt) sacc[nt] = (v8f)0.0f;
        #pragma unroll
        for (int s = 0; s < 2; ++s) {
            #pragma unroll
            for (int nt = 0; nt < 4; ++nt) {
                const _Float16* p = &lK[buf][(nt * 16 + ln) * 64 + s * 32 + hi * 8];
                sacc[nt] = wmma_f16(qa[s], frag16(p, p + 16), sacc[nt]);
            }
        }

        // ---- online softmax ----
        #pragma unroll
        for (int j = 0; j < 8; ++j) {
            float mx = -1e30f;
            #pragma unroll
            for (int nt = 0; nt < 4; ++nt) {
                sacc[nt][j] *= scale;
                mx = fmaxf(mx, sacc[nt][j]);
            }
            mx = redmax16(mx);
            float newm = fmaxf(rowmax[j], mx);
            float corr = __expf(rowmax[j] - newm);
            rowmax[j] = newm;
            float psum = 0.0f;
            #pragma unroll
            for (int nt = 0; nt < 4; ++nt) {
                float p = __expf(sacc[nt][j] - newm);
                lP[w][(hi * 8 + j) * 64 + nt * 16 + ln] = (_Float16)p;
                psum += p;
            }
            psum = redsum16(psum);
            rowsum[j] = rowsum[j] * corr + psum;
            #pragma unroll
            for (int nt = 0; nt < 4; ++nt) oacc[nt][j] *= corr;
        }

        // ---- O += P @ V ----
        #pragma unroll
        for (int s = 0; s < 2; ++s) {
            const _Float16* pa = &lP[w][ln * 64 + s * 32 + hi * 8];
            v16h fa = frag16(pa, pa + 16);
            #pragma unroll
            for (int nt = 0; nt < 4; ++nt) {
                const _Float16* p = &lV[buf][(nt * 16 + ln) * 64 + s * 32 + hi * 8];
                oacc[nt] = wmma_f16(fa, frag16(p, p + 16), oacc[nt]);
            }
        }
#ifdef HAS_TDM
        __syncthreads();
        if (w == 0 && kt + 2 < nk) {
            tdm_load_2d((unsigned)(size_t)&lK[buf][0],
                        kbase + (size_t)(b * 1024 + (kt + 2) * 64) * 1536, 64, 64, 1536);
            tdm_load_2d((unsigned)(size_t)&lV[buf][0],
                        vbase + (kt + 2) * 64, 64, 64, 1024);
        }
#endif
    }

    // ---- normalize, store f16 [row][h*64+d] ----
    #pragma unroll
    for (int j = 0; j < 8; ++j) {
        float inv = 1.0f / rowsum[j];
        int row = r0 + w * 16 + hi * 8 + j;
        _Float16* dst = attnout + (size_t)(b * 1024 + row) * 768 + h * 64;
        #pragma unroll
        for (int nt = 0; nt < 4; ++nt)
            dst[nt * 16 + ln] = (_Float16)(oacc[nt][j] * inv);
    }
}

// ---------------------------------------------------------------------------
// Host-side launch
// ---------------------------------------------------------------------------
extern "C" void kernel_launch(void* const* d_in, const int* in_sizes, int n_in,
                              void* d_out, int out_size, void* d_ws, size_t ws_size,
                              hipStream_t stream) {
    const float* x     = (const float*)d_in[0];   // [8,1024,768]
    const float* Wqkv  = (const float*)d_in[1];   // [768,2304]
    const float* bqkv  = (const float*)d_in[2];   // [2304]
    const float* Wproj = (const float*)d_in[3];   // [768,768]
    const float* bproj = (const float*)d_in[4];   // [768]
    float* out = (float*)d_out;                   // [8,1024,768]

    constexpr int M  = 8192;      // B*N
    constexpr int C  = 768;
    constexpr int C3 = 2304;

    // workspace carve (f16): x | WqkvT | WprojT | qk | vT | attn  (~67 MB)
    _Float16* xh     = (_Float16*)d_ws;
    _Float16* WqkvT  = xh + (size_t)M * C;                 // [2304][768]
    _Float16* WprojT = WqkvT + (size_t)C3 * C;             // [768][768]
    _Float16* qkh    = WprojT + (size_t)C * C;             // [8192][1536]
    _Float16* vTh    = qkh + (size_t)M * 1536;             // [96*64][1024]
    _Float16* attnh  = vTh + (size_t)96 * 64 * 1024;       // [8192][768]

    {
        int n1 = M * C, n2 = C * C3, n3 = C * C;
        cvt_f32_f16<<<(n1 + 255) / 256, 256, 0, stream>>>(x, xh, n1);
        cvt_transpose_f16<<<(n2 + 255) / 256, 256, 0, stream>>>(Wqkv, WqkvT, C, C3);
        cvt_transpose_f16<<<(n3 + 255) / 256, 256, 0, stream>>>(Wproj, WprojT, C, C);
    }

    // qkv = x @ Wqkv + bqkv  (q,k -> qkh; v -> vTh transposed)
    gemm_wmma<true><<<dim3(C3 / 128, M / 128), 256, 0, stream>>>(
        xh, WqkvT, bqkv, (void*)qkh, vTh, M, C3, C);

    // flash attention -> attnh
    attn_flash<<<dim3(16, 96), 128, 0, stream>>>(qkh, vTh, attnh);

    // out = attn @ Wproj + bproj (f32)
    gemm_wmma<false><<<dim3(C / 128, M / 128), 256, 0, stream>>>(
        attnh, WprojT, bproj, (void*)out, nullptr, M, C, C);
}
